// GAT_58445914964485
// MI455X (gfx1250) — matrix operand order repaired and checked by reading, not compiled, not run on hardware
//
#include <hip/hip_runtime.h>
#include <hip/hip_fp16.h>

typedef __attribute__((ext_vector_type(16))) _Float16 v16h;
typedef __attribute__((ext_vector_type(8)))  float    v8f;

#define N_NODES 50000
#define N_EDGES 800000
#define DIN     128
#define NHEADS  4

// ---------------- f32 -> f16 convert (activations) ----------------
__global__ void k_convert_f16(const float* __restrict__ s, _Float16* __restrict__ d, int n) {
    int i = blockIdx.x * blockDim.x + threadIdx.x;
    if (i < n) d[i] = (_Float16)s[i];
}

// W [128, ncols] row-major f32 -> Wt [ncols, 128] f16 (column-major-of-W, contiguous K)
__global__ void k_transpose_w(const float* __restrict__ W, _Float16* __restrict__ Wt, int ncols) {
    int i = blockIdx.x * blockDim.x + threadIdx.x;
    int total = ncols * DIN;
    if (i >= total) return;
    int n = i / DIN, k = i % DIN;
    Wt[n * DIN + k] = (_Float16)W[k * ncols + n];
}

// ---------------- WMMA GEMM: feat[N,ncols] = hf16[N,128] @ W ----------------
// One 16x16 C tile per wave; K=128 = 4 steps of 32 via v_wmma_f32_16x16x32_f16.
__global__ void k_gemm_wmma(const _Float16* __restrict__ A,   // [N_NODES,128] row-major
                            const _Float16* __restrict__ Bt,  // [ncols,128]  (W transposed)
                            float* __restrict__ C, int ncols) {
    const int nTilesN = ncols >> 4;
    const int wave = (int)((blockIdx.x * blockDim.x + threadIdx.x) >> 5);
    const int lane = threadIdx.x & 31;
    const int tm = wave / nTilesN;
    const int tn = wave % nTilesN;
    if (tm >= (N_NODES >> 4)) return;              // wave-uniform guard (EXEC all-1 at WMMA)

    const int ml = lane & 15;                      // row within A tile / col within C tile
    const int hi = lane >> 4;                      // half-wave select
    const _Float16* arow = A  + (size_t)(tm * 16 + ml) * DIN;
    const _Float16* brow = Bt + (size_t)(tn * 16 + ml) * DIN;

    v8f c = {};
#pragma unroll
    for (int kk = 0; kk < 4; ++kk) {
        union { v16h v; uint4 q[2]; } a, b;
        // A 16x32 f16 layout: lane(ml,hi) holds K = hi*16 + j (j=0..15), contiguous
        a.q[0] = *(const uint4*)(arow + kk * 32 + hi * 16);
        a.q[1] = *(const uint4*)(arow + kk * 32 + hi * 16 + 8);
        // B 32x16 f16 layout: lane(n=ml,hi) holds K = hi*16 + j (j=0..15), contiguous in Wt
        b.q[0] = *(const uint4*)(brow + kk * 32 + hi * 16);
        b.q[1] = *(const uint4*)(brow + kk * 32 + hi * 16 + 8);
        c = __builtin_amdgcn_wmma_f32_16x16x32_f16(false, a.v, false, b.v,
                                                   (short)0, c, false, false);
    }
    // C layout: VGPR r, lane(ml,hi) -> C[hi*8+r][ml]
    float* crow = C + (size_t)(tm * 16 + hi * 8) * ncols + tn * 16 + ml;
#pragma unroll
    for (int r = 0; r < 8; ++r) crow[(size_t)r * ncols] = c[r];
}

// ---------------- el/er: per (node, head) dot(feat, al/ar) ----------------
__global__ void k_eler(const float* __restrict__ feat, const float* __restrict__ al,
                       const float* __restrict__ ar, float* __restrict__ el,
                       float* __restrict__ er, int F) {
    int i = blockIdx.x * blockDim.x + threadIdx.x;
    if (i >= N_NODES * NHEADS) return;
    int n = i / NHEADS, h = i % NHEADS;
    const float* fr = feat + (size_t)n * NHEADS * F + (size_t)h * F;
    const float* a  = al + h * F;
    const float* b  = ar + h * F;
    float sl = 0.f, sr = 0.f;
    for (int f = 0; f < F; ++f) { float v = fr[f]; sl += v * a[f]; sr += v * b[f]; }
    el[i] = sl; er[i] = sr;
}

// ---------------- init: m=-inf, denom=0, agg=bias ----------------
__global__ void k_init(float* __restrict__ m, float* __restrict__ denom,
                       float* __restrict__ agg, const float* __restrict__ bias, int HF) {
    int i = blockIdx.x * blockDim.x + threadIdx.x;
    if (i < N_NODES * NHEADS) { m[i] = __int_as_float(0xFF800000); denom[i] = 0.f; }
    if (i < N_NODES * HF)     { agg[i] = bias[i % HF]; }
}

// ---------------- pass A: e = leaky_relu(el[src]+er[dst]); seg-max into m[dst] -----
__global__ void k_edge_logit(const int* __restrict__ src, const int* __restrict__ dst,
                             const float* __restrict__ el, const float* __restrict__ er,
                             float* __restrict__ eBuf, float* __restrict__ m) {
    int i = blockIdx.x * blockDim.x + threadIdx.x;
    if (i >= N_EDGES * NHEADS) return;
    int e = i >> 2, h = i & 3;
    int s = src[e], d = dst[e];
    float v = el[s * NHEADS + h] + er[d * NHEADS + h];
    v = v > 0.f ? v : 0.2f * v;
    eBuf[i] = v;
    float* addr = m + d * NHEADS + h;           // float atomic-max via bit trick
    if (v >= 0.f) atomicMax((int*)addr, __float_as_int(v));
    else          atomicMin((unsigned int*)addr, __float_as_uint(v));
}

// ---------------- pass B: ex = exp(e - m[dst]); seg-sum into denom ----------------
__global__ void k_edge_exp(const int* __restrict__ dst, float* __restrict__ eBuf,
                           const float* __restrict__ m, float* __restrict__ denom) {
    int i = blockIdx.x * blockDim.x + threadIdx.x;
    if (i >= N_EDGES * NHEADS) return;
    int e = i >> 2, h = i & 3;
    int d = dst[e];
    float ex = __expf(eBuf[i] - m[d * NHEADS + h]);
    eBuf[i] = ex;
    atomicAdd(denom + d * NHEADS + h, ex);
}

// ---------------- pass C: agg[dst] += feat[src] * alpha ----------------
__global__ void k_edge_agg(const int* __restrict__ src, const int* __restrict__ dst,
                           const float* __restrict__ feat, const float* __restrict__ eBuf,
                           const float* __restrict__ denom, float* __restrict__ agg, int F) {
    const int HF4 = NHEADS * F / 4;             // float4 chunks per edge
    long long i = (long long)blockIdx.x * blockDim.x + threadIdx.x;
    if (i >= (long long)N_EDGES * HF4) return;
    int e = (int)(i / HF4), c = (int)(i % HF4);
    int h = (c * 4) / F;
    int s = src[e], d = dst[e];
    float dn = denom[d * NHEADS + h];
    float alpha = eBuf[e * NHEADS + h] / fmaxf(dn, 1e-9f);
    const float4 fv = *(const float4*)(feat + (size_t)s * NHEADS * F + (size_t)c * 4);
    float* o = agg + (size_t)d * NHEADS * F + (size_t)c * 4;
    atomicAdd(o + 0, fv.x * alpha);
    atomicAdd(o + 1, fv.y * alpha);
    atomicAdd(o + 2, fv.z * alpha);
    atomicAdd(o + 3, fv.w * alpha);
}

// ---------------- epilogues ----------------
__global__ void k_finish_elu(const float* __restrict__ agg, _Float16* __restrict__ hf16, int HF) {
    int i = blockIdx.x * blockDim.x + threadIdx.x;
    if (i >= N_NODES * HF) return;
    float v = agg[i];
    v = v > 0.f ? v : (__expf(v) - 1.f);        // ELU(alpha=1)
    hf16[i] = (_Float16)v;
}

__global__ void k_finish_mean(const float* __restrict__ agg, float* __restrict__ out) {
    int i = blockIdx.x * blockDim.x + threadIdx.x;
    if (i >= N_NODES * 64) return;
    int n = i >> 6, f = i & 63;
    const float* r = agg + (size_t)n * 256;
    out[i] = 0.25f * (r[f] + r[64 + f] + r[128 + f] + r[192 + f]);
}

// ---------------- driver ----------------
extern "C" void kernel_launch(void* const* d_in, const int* in_sizes, int n_in,
                              void* d_out, int out_size, void* d_ws, size_t ws_size,
                              hipStream_t stream) {
    const float* x   = (const float*)d_in[0];
    const int*   src = (const int*)  d_in[1];
    const int*   dst = (const int*)  d_in[2];
    const float *W[4], *al[4], *ar[4], *bs[4];
    for (int L = 0; L < 4; ++L) {
        W[L]  = (const float*)d_in[3 + 4 * L + 0];
        al[L] = (const float*)d_in[3 + 4 * L + 1];
        ar[L] = (const float*)d_in[3 + 4 * L + 2];
        bs[L] = (const float*)d_in[3 + 4 * L + 3];
    }

    char* ws = (char*)d_ws;
    size_t off = 0;
    _Float16* hf16 = (_Float16*)(ws + off); off += (size_t)N_NODES * 128 * 2;   // 12.8 MB
    _Float16* Wt   = (_Float16*)(ws + off); off += (size_t)256 * 128 * 2;       // 64 KB
    float* feat  = (float*)(ws + off); off += (size_t)N_NODES * 256 * 4;        // 51.2 MB
    float* agg   = (float*)(ws + off); off += (size_t)N_NODES * 256 * 4;        // 51.2 MB
    float* el    = (float*)(ws + off); off += (size_t)N_NODES * NHEADS * 4;
    float* er    = (float*)(ws + off); off += (size_t)N_NODES * NHEADS * 4;
    float* mbuf  = (float*)(ws + off); off += (size_t)N_NODES * NHEADS * 4;
    float* denom = (float*)(ws + off); off += (size_t)N_NODES * NHEADS * 4;
    float* eBuf  = (float*)(ws + off); off += (size_t)N_EDGES * NHEADS * 4;     // 12.8 MB

    const int B = 256;
    // x -> f16 once
    {
        int n = N_NODES * 128;
        k_convert_f16<<<(n + B - 1) / B, B, 0, stream>>>(x, hf16, n);
    }

    for (int L = 0; L < 4; ++L) {
        const int ncols = (L == 3) ? 256 : 128;   // H*F
        const int F     = (L == 3) ? 64  : 32;

        k_transpose_w<<<(ncols * DIN + B - 1) / B, B, 0, stream>>>(W[L], Wt, ncols);

        int tiles = (N_NODES / 16) * (ncols / 16);        // waves needed
        k_gemm_wmma<<<(tiles + 7) / 8, B, 0, stream>>>(hf16, Wt, feat, ncols);

        k_eler<<<(N_NODES * NHEADS + B - 1) / B, B, 0, stream>>>(feat, al[L], ar[L], el, er, F);

        int ninit = N_NODES * ncols;
        k_init<<<(ninit + B - 1) / B, B, 0, stream>>>(mbuf, denom, agg, bs[L], ncols);

        int nEH = N_EDGES * NHEADS;
        k_edge_logit<<<(nEH + B - 1) / B, B, 0, stream>>>(src, dst, el, er, eBuf, mbuf);
        k_edge_exp<<<(nEH + B - 1) / B, B, 0, stream>>>(dst, eBuf, mbuf, denom);

        long long nagg = (long long)N_EDGES * (NHEADS * F / 4);
        k_edge_agg<<<(unsigned)((nagg + B - 1) / B), B, 0, stream>>>(src, dst, feat, eBuf,
                                                                     denom, agg, F);
        if (L < 3) {
            int n = N_NODES * ncols;
            k_finish_elu<<<(n + B - 1) / B, B, 0, stream>>>(agg, hf16, ncols);
        } else {
            int n = N_NODES * 64;
            k_finish_mean<<<(n + B - 1) / B, B, 0, stream>>>(agg, (float*)d_out);
        }
    }
}
